// MultiHeadAttention_11871289606603
// MI455X (gfx1250) — compile-verified, matrix operand order checked by reading, not run on hardware
//
#include <hip/hip_runtime.h>
#include <hip/hip_bf16.h>

typedef __bf16 v16bf __attribute__((ext_vector_type(16)));
typedef __bf16 v8bf  __attribute__((ext_vector_type(8)));
typedef float  v8f   __attribute__((ext_vector_type(8)));
typedef int    v4i   __attribute__((ext_vector_type(4)));

#define D_MODEL 1024
#define NHEAD   16
#define DKH     64
#define SEQ     2048
#define BATCH   4
#define MROWS   (BATCH * SEQ)   // 8192

#define BM 128
#define BN 128
#define BK 32
#define QB 128
#define KB 64

// ---- CDNA5 async global->LDS staging (ASYNCcnt), guarded so either toolchain builds
#if defined(__has_builtin)
#if __has_builtin(__builtin_amdgcn_global_load_async_to_lds_b128) && \
    __has_builtin(__builtin_amdgcn_s_wait_asynccnt)
#define USE_ASYNC_LDS 1
#endif
#endif
#ifndef USE_ASYNC_LDS
#define USE_ASYNC_LDS 0
#endif

static __device__ __forceinline__ void async_copy16(const __bf16* g, __bf16* l) {
#if USE_ASYNC_LDS
    // prototype (from hipcc diagnostic): (v4i __device__*, v4i __shared__*, imm, imm)
    __builtin_amdgcn_global_load_async_to_lds_b128(
        (__attribute__((address_space(1))) v4i*)g,
        (__attribute__((address_space(3))) v4i*)l, 0, 0);
#else
    *(v8bf*)l = *(const v8bf*)g;
#endif
}
static __device__ __forceinline__ void async_wait() {
#if USE_ASYNC_LDS
    __builtin_amdgcn_s_wait_asynccnt(0);
#endif
}

static __device__ __forceinline__ __bf16 f32_to_bf16(float f) {
    unsigned u = __builtin_bit_cast(unsigned, f);
    u += 0x7FFFu + ((u >> 16) & 1u);                 // round-to-nearest-even
    unsigned short h = (unsigned short)(u >> 16);
    return __builtin_bit_cast(__bf16, h);
}

// 16x32 A-fragment / 32x16 B-fragment loader (B's columns = contiguous rows of the
// untransposed [N,K] source). ISA 7.12.2: lanes 0-15 hold row(lane), K {0..7,16..23};
// lanes 16-31 hold the same rows, K {8..15,24..31}.
static __device__ __forceinline__ v16bf load_frag(const __bf16* base, int row,
                                                  int ldm, int k0, int lane) {
    const int half = lane >> 4;
    const __bf16* p = base + (size_t)(row + (lane & 15)) * ldm + k0 + 8 * half;
    v8bf lo = *(const v8bf*)p;
    v8bf hi = *(const v8bf*)(p + 16);
    return __builtin_shufflevector(lo, hi, 0, 1, 2, 3, 4, 5, 6, 7,
                                   8, 9, 10, 11, 12, 13, 14, 15);
}

// ---------------------------------------------------------------- convert
__global__ __launch_bounds__(256) void cvt_f32_bf16(const float* __restrict__ x,
                                                    __bf16* __restrict__ y, int n) {
    int i = (blockIdx.x * 256 + threadIdx.x) * 4;
    if (i < n) {
        float4 f = *(const float4*)(x + i);
        unsigned s0 = (unsigned)__builtin_bit_cast(unsigned short, f32_to_bf16(f.x));
        unsigned s1 = (unsigned)__builtin_bit_cast(unsigned short, f32_to_bf16(f.y));
        unsigned s2 = (unsigned)__builtin_bit_cast(unsigned short, f32_to_bf16(f.z));
        unsigned s3 = (unsigned)__builtin_bit_cast(unsigned short, f32_to_bf16(f.w));
        uint2 pk;
        pk.x = s0 | (s1 << 16);
        pk.y = s2 | (s3 << 16);
        *(uint2*)(y + i) = pk;
    }
}

// ---------------------------------------------------------------- GEMM
// C[M,N] = A[M,K] * Bw[N,K]^T. 128x128 block, 8 waves as 4x2 grid of 32x64 tiles:
// 8 WMMAs per k-step from 2 A-frags x 4 B-frags. Double-buffered async LDS staging.
// MODE 0: f32 row-major out.  MODE 1: bf16 out scattered to [B,H,T,DKH].
template <int MODE>
__global__ __launch_bounds__(256) void gemm_bf16_wmma(
    const __bf16* __restrict__ A, const __bf16* __restrict__ Bw,
    float* __restrict__ Cf, __bf16* __restrict__ Ch, int Kdim, int Ndim) {
    __shared__ __bf16 As[2][BM][BK + 8];
    __shared__ __bf16 Bs[2][BN][BK + 8];

    const int tid  = threadIdx.x;
    const int wave = tid >> 5;
    const int lane = tid & 31;
    const int wr   = wave >> 1;      // 0..3 : 32-row slice
    const int wc   = wave & 1;       // 0..1 : 64-col slice
    const int m0   = blockIdx.x * BM;
    const int n0   = blockIdx.y * BN;

    v8f acc[2][4] = {};

    auto stage = [&](int buf, int kk) {
        int idx = tid * 8;
#pragma unroll
        for (int it = 0; it < 2; ++it) {
            int r = idx >> 5, c = idx & 31;
            async_copy16(A  + (size_t)(m0 + r) * Kdim + kk + c, &As[buf][r][c]);
            async_copy16(Bw + (size_t)(n0 + r) * Kdim + kk + c, &Bs[buf][r][c]);
            idx += 2048;
        }
    };

    stage(0, 0);
    async_wait();
    __syncthreads();

    int buf = 0;
    for (int kk = 0; kk < Kdim; kk += BK) {
        if (kk + BK < Kdim) {
            stage(buf ^ 1, kk + BK);            // overlap next tile with compute
            __builtin_prefetch(A + (size_t)(m0 + (tid >> 1)) * Kdim + kk + 2 * BK, 0, 1);
        }
        v16bf a0 = load_frag(&As[buf][0][0], wr * 32,      BK + 8, 0, lane);
        v16bf a1 = load_frag(&As[buf][0][0], wr * 32 + 16, BK + 8, 0, lane);
#pragma unroll
        for (int nb = 0; nb < 4; ++nb) {
            v16bf bfr = load_frag(&Bs[buf][0][0], wc * 64 + nb * 16, BK + 8, 0, lane);
            acc[0][nb] = __builtin_amdgcn_wmma_f32_16x16x32_bf16(
                false, a0, false, bfr, (short)0, acc[0][nb], false, false);
            acc[1][nb] = __builtin_amdgcn_wmma_f32_16x16x32_bf16(
                false, a1, false, bfr, (short)0, acc[1][nb], false, false);
        }
        async_wait();
        __syncthreads();
        buf ^= 1;
    }

    const int half = lane >> 4;
    const int cl   = lane & 15;
#pragma unroll
    for (int fa = 0; fa < 2; ++fa) {
#pragma unroll
        for (int nb = 0; nb < 4; ++nb) {
            int n = n0 + wc * 64 + nb * 16 + cl;
#pragma unroll
            for (int j = 0; j < 8; ++j) {
                int   m = m0 + wr * 32 + fa * 16 + j + 8 * half;
                float v = acc[fa][nb][j];
                if (MODE == 0) {
                    Cf[(size_t)m * Ndim + n] = v;
                } else {
                    int b = m / SEQ, t = m % SEQ;
                    int h = n >> 6, d = n & 63;
                    Ch[(((size_t)(b * NHEAD + h)) * SEQ + t) * DKH + d] =
                        f32_to_bf16(v);
                }
            }
        }
    }
}

// ---------------------------------------------------------------- flash attention
__global__ __launch_bounds__(256) void flash_attn(
    const __bf16* __restrict__ Qh, const __bf16* __restrict__ Kh,
    const __bf16* __restrict__ Vh, __bf16* __restrict__ Ob) {
    __shared__ __bf16 Kt[KB][DKH + 8];        // [key][dk]
    __shared__ __bf16 Vt[DKH][KB + 8];        // transposed: [dk][key]
    __shared__ __bf16 Pb[8][16][KB + 8];      // per-wave P bounce buffer

    const int tid  = threadIdx.x;
    const int wave = tid >> 5;
    const int lane = tid & 31;
    const int half = lane >> 4;
    const int cl   = lane & 15;

    const int qt = blockIdx.x % (SEQ / QB);
    const int bh = blockIdx.x / (SEQ / QB);   // = b*NHEAD + h
    const int b  = bh / NHEAD;
    const int h  = bh % NHEAD;

    const int    q0       = qt * QB + wave * 16;
    const size_t qkv_base = (size_t)bh * SEQ * DKH;

    v16bf qf[2];
#pragma unroll
    for (int ks = 0; ks < 2; ++ks)
        qf[ks] = load_frag(Qh + qkv_base, q0, DKH, ks * 32, lane);

    float m_st[8], l_st[8];
#pragma unroll
    for (int j = 0; j < 8; ++j) { m_st[j] = -3.0e38f; l_st[j] = 0.0f; }
    v8f acc[4] = {};

    const int nkb = (qt + 1) * (QB / KB);     // causal: only key blocks <= q tile
    for (int kb_i = 0; kb_i < nkb; ++kb_i) {
        const int kb = kb_i * KB;
        // cooperative stage: K via async engine, V transposed through registers
        int idx = tid * 8;
#pragma unroll
        for (int it = 0; it < 2; ++it) {
            int r = idx >> 6, c = idx & 63;
            async_copy16(Kh + qkv_base + (size_t)(kb + r) * DKH + c, &Kt[r][c]);
            v8bf vv = *(const v8bf*)(Vh + qkv_base + (size_t)(kb + r) * DKH + c);
#pragma unroll
            for (int e = 0; e < 8; ++e) Vt[c + e][r] = vv[e];
            idx += 2048;
        }
        async_wait();
        __syncthreads();

        if (kb <= q0 + 15) {
            // ---- S = (Q K^T) * 1/sqrt(dk), causal-masked
            v8f s[4];
#pragma unroll
            for (int nb = 0; nb < 4; ++nb) {
                v8f sa = {};
#pragma unroll
                for (int ks = 0; ks < 2; ++ks) {
                    v16bf kf = load_frag(&Kt[0][0], nb * 16, DKH + 8, ks * 32, lane);
                    sa = __builtin_amdgcn_wmma_f32_16x16x32_bf16(
                        false, qf[ks], false, kf, (short)0, sa, false, false);
                }
                s[nb] = sa;
            }
#pragma unroll
            for (int nb = 0; nb < 4; ++nb) {
                int kg = kb + nb * 16 + cl;
#pragma unroll
                for (int j = 0; j < 8; ++j) {
                    int qg = q0 + j + 8 * half;
                    float v = s[nb][j] * 0.125f;         // 1/sqrt(64)
                    s[nb][j] = (kg <= qg) ? v : -3.0e38f;
                }
            }
            // ---- online softmax (row reductions across 16-lane halves)
            float mnew[8], scl[8];
#pragma unroll
            for (int j = 0; j < 8; ++j) {
                float mx = fmaxf(fmaxf(s[0][j], s[1][j]), fmaxf(s[2][j], s[3][j]));
#pragma unroll
                for (int off = 1; off < 16; off <<= 1)
                    mx = fmaxf(mx, __shfl_xor(mx, off, 32));
                mnew[j] = fmaxf(m_st[j], mx);
                scl[j]  = __expf(m_st[j] - mnew[j]);
            }
#pragma unroll
            for (int nb = 0; nb < 4; ++nb)
#pragma unroll
                for (int j = 0; j < 8; ++j)
                    s[nb][j] = __expf(s[nb][j] - mnew[j]);
#pragma unroll
            for (int j = 0; j < 8; ++j) {
                float rs = s[0][j] + s[1][j] + s[2][j] + s[3][j];
#pragma unroll
                for (int off = 1; off < 16; off <<= 1)
                    rs += __shfl_xor(rs, off, 32);
                l_st[j] = l_st[j] * scl[j] + rs;
                m_st[j] = mnew[j];
            }
#pragma unroll
            for (int nb = 0; nb < 4; ++nb)
#pragma unroll
                for (int j = 0; j < 8; ++j) acc[nb][j] *= scl[j];

            // ---- P: C-layout -> A-layout via per-wave LDS bounce
#pragma unroll
            for (int nb = 0; nb < 4; ++nb)
#pragma unroll
                for (int j = 0; j < 8; ++j)
                    Pb[wave][j + 8 * half][nb * 16 + cl] = f32_to_bf16(s[nb][j]);

            // ---- acc += P @ V  (contraction over keys, 2 x K=32)
#pragma unroll
            for (int ks = 0; ks < 2; ++ks) {
                v16bf pf = load_frag(&Pb[wave][0][0], 0, KB + 8, ks * 32, lane);
#pragma unroll
                for (int nb = 0; nb < 4; ++nb) {
                    v16bf vf = load_frag(&Vt[0][0], nb * 16, KB + 8, ks * 32, lane);
                    acc[nb] = __builtin_amdgcn_wmma_f32_16x16x32_bf16(
                        false, pf, false, vf, (short)0, acc[nb], false, false);
                }
            }
        }
        __syncthreads();
    }

    // epilogue: normalize, write bf16 attn output in [B,T,D] layout
#pragma unroll
    for (int j = 0; j < 8; ++j) {
        float inv = 1.0f / l_st[j];
        int   t   = q0 + j + 8 * half;
#pragma unroll
        for (int nb = 0; nb < 4; ++nb) {
            int d = nb * 16 + cl;
            Ob[((size_t)b * SEQ + t) * D_MODEL + h * DKH + d] =
                f32_to_bf16(acc[nb][j] * inv);
        }
    }
}

// ---------------------------------------------------------------- launch
extern "C" void kernel_launch(void* const* d_in, const int* in_sizes, int n_in,
                              void* d_out, int out_size, void* d_ws, size_t ws_size,
                              hipStream_t stream) {
    (void)in_sizes; (void)n_in; (void)out_size; (void)ws_size;
    const float* X  = (const float*)d_in[0];
    const float* Wq = (const float*)d_in[1];
    const float* Wk = (const float*)d_in[2];
    const float* Wv = (const float*)d_in[3];
    const float* Wo = (const float*)d_in[4];
    float* out = (float*)d_out;

    char*  ws = (char*)d_ws;
    size_t o  = 0;
    __bf16* Xb  = (__bf16*)(ws + o); o += (size_t)MROWS * D_MODEL * 2;
    __bf16* Wqb = (__bf16*)(ws + o); o += (size_t)D_MODEL * D_MODEL * 2;
    __bf16* Wkb = (__bf16*)(ws + o); o += (size_t)D_MODEL * D_MODEL * 2;
    __bf16* Wvb = (__bf16*)(ws + o); o += (size_t)D_MODEL * D_MODEL * 2;
    __bf16* Wob = (__bf16*)(ws + o); o += (size_t)D_MODEL * D_MODEL * 2;
    __bf16* Qh  = (__bf16*)(ws + o); o += (size_t)MROWS * D_MODEL * 2;
    __bf16* Kh  = (__bf16*)(ws + o); o += (size_t)MROWS * D_MODEL * 2;
    __bf16* Vh  = (__bf16*)(ws + o); o += (size_t)MROWS * D_MODEL * 2;
    __bf16* Ab  = (__bf16*)(ws + o); o += (size_t)MROWS * D_MODEL * 2;

    const int nX = MROWS * D_MODEL;          // 8388608
    const int nW = D_MODEL * D_MODEL;        // 1048576
    cvt_f32_bf16<<<nX / 1024, 256, 0, stream>>>(X,  Xb,  nX);
    cvt_f32_bf16<<<nW / 1024, 256, 0, stream>>>(Wq, Wqb, nW);
    cvt_f32_bf16<<<nW / 1024, 256, 0, stream>>>(Wk, Wkb, nW);
    cvt_f32_bf16<<<nW / 1024, 256, 0, stream>>>(Wv, Wvb, nW);
    cvt_f32_bf16<<<nW / 1024, 256, 0, stream>>>(Wo, Wob, nW);

    dim3 gg(MROWS / BM, D_MODEL / BN);       // 64 x 8
    gemm_bf16_wmma<1><<<gg, 256, 0, stream>>>(Xb, Wqb, nullptr, Qh, D_MODEL, D_MODEL);
    gemm_bf16_wmma<1><<<gg, 256, 0, stream>>>(Xb, Wkb, nullptr, Kh, D_MODEL, D_MODEL);
    gemm_bf16_wmma<1><<<gg, 256, 0, stream>>>(Xb, Wvb, nullptr, Vh, D_MODEL, D_MODEL);

    flash_attn<<<BATCH * NHEAD * (SEQ / QB), 256, 0, stream>>>(Qh, Kh, Vh, Ab);

    gemm_bf16_wmma<0><<<gg, 256, 0, stream>>>(Ab, Wob, out, nullptr, D_MODEL, D_MODEL);
}
